// BidirectionalAttention_12343736009469
// MI455X (gfx1250) — matrix operand order repaired and checked by reading, not compiled, not run on hardware
//
#include <hip/hip_runtime.h>

// MI455X / gfx1250, wave32. bf16 WMMA (f32 accumulate) attention block.
// B=2, N=2048, dim=2048, H=16, Dh=128. M = B*N = 4096.
// Global->LDS tile staging uses CDNA5 GLOBAL_LOAD_ASYNC_TO_LDS_B128 (ASYNCcnt).

typedef __bf16 bf16;
typedef __attribute__((ext_vector_type(8)))  bf16  v8bf;
typedef __attribute__((ext_vector_type(16))) bf16  v16bf;
typedef __attribute__((ext_vector_type(8)))  float v8f;

__device__ __forceinline__ bf16 f2bf(float f) {
  unsigned u = __builtin_bit_cast(unsigned, f);
  unsigned r = u + 0x7FFFu + ((u >> 16) & 1u);   // round-to-nearest-even
  unsigned short h = (unsigned short)(r >> 16);
  return __builtin_bit_cast(bf16, h);
}

__device__ __forceinline__ v8f vzero8() {
  v8f z;
#pragma unroll
  for (int i = 0; i < 8; i++) z[i] = 0.0f;
  return z;
}

__device__ __forceinline__ v16bf cat16(v8bf lo, v8bf hi) {
  return __builtin_shufflevector(lo, hi, 0,1,2,3,4,5,6,7,8,9,10,11,12,13,14,15);
}

// Low 32 bits of the generic (flat) shared-aperture address are the LDS byte
// offset (ISA 10.2).
__device__ __forceinline__ unsigned lds_off(const void* p) {
  return (unsigned)(unsigned long long)(uintptr_t)p;
}

// CDNA5 async DMA: 16B per lane, global -> LDS, tracked by ASYNCcnt.
__device__ __forceinline__ void async_b128(unsigned ldsoff, const void* gptr) {
  asm volatile("global_load_async_to_lds_b128 %0, %1, off"
               :: "v"(ldsoff), "v"((unsigned long long)(uintptr_t)gptr)
               : "memory");
}

__device__ __forceinline__ void wait_async0() {
  asm volatile("s_wait_asynccnt 0" ::: "memory");
}

// A-fragment (16x32 bf16, MxK). Per ISA: lanes 0-15 hold row M=lane,
// K = {0..7, 16..23}; lanes 16-31 hold row M=lane-16, K = {8..15, 24..31}.
// base points at the tile origin (row-major, k contiguous), ld in elements.
__device__ __forceinline__ v16bf load_a_frag(const bf16* base, int ld) {
  const int lane = threadIdx.x & 31;
  const bf16* p = base + (lane & 15) * ld + ((lane >> 4) << 3);
  v8bf lo = *(const v8bf*)(p);
  v8bf hi = *(const v8bf*)(p + 16);
  return cat16(lo, hi);
}

// B-fragment (32x16 bf16, KxN). Lanes 0-15: col N=lane, K={0..7, 16..23};
// lanes 16-31: col N=lane-16, K={8..15, 24..31}. base = [n][k] layout (k contiguous).
__device__ __forceinline__ v16bf load_b_frag(const bf16* base, int ld) {
  const int lane = threadIdx.x & 31;
  const bf16* p = base + (lane & 15) * ld + ((lane >> 4) << 3);
  v8bf lo = *(const v8bf*)(p);
  v8bf hi = *(const v8bf*)(p + 16);
  return cat16(lo, hi);
}

__device__ __forceinline__ v8f wmma_bf(v16bf a, v16bf b, v8f c) {
  return __builtin_amdgcn_wmma_f32_16x16x32_bf16(false, a, false, b,
                                                 (short)0, c, false, false);
}

// ---------------------------------------------------------------- prep ----
__global__ void __launch_bounds__(256) cast_f32_to_bf16(
    const float* __restrict__ in, bf16* __restrict__ out, int n) {
  int i = (blockIdx.x * 256 + threadIdx.x) * 4;
  if (i < n) {
    float4 v = *(const float4*)(in + i);
    out[i + 0] = f2bf(v.x);
    out[i + 1] = f2bf(v.y);
    out[i + 2] = f2bf(v.z);
    out[i + 3] = f2bf(v.w);
  }
}

// in: [K][C] f32 row-major -> out: [C][K] bf16 row-major (tiled via LDS).
__global__ void __launch_bounds__(256) transpose_cast(
    const float* __restrict__ in, bf16* __restrict__ out, int K, int C) {
  __shared__ float t[32][33];
  const int tx = threadIdx.x & 31, ty = threadIdx.x >> 5;
  const int c0 = blockIdx.x * 32, k0 = blockIdx.y * 32;
#pragma unroll
  for (int i = 0; i < 4; i++)
    t[ty + i * 8][tx] = in[(size_t)(k0 + ty + i * 8) * C + c0 + tx];
  __syncthreads();
#pragma unroll
  for (int i = 0; i < 4; i++)
    out[(size_t)(c0 + ty + i * 8) * K + k0 + tx] = f2bf(t[tx][ty + i * 8]);
}

// ------------------------------------------------- GEMM1: qkv + RoPE ------
// C[4096 x 6144] = X[4096 x 2048] * WqkvT^T ; epilogue applies RoPE to q,k
// and scatters into Q[b,h,n,d], K[b,h,n,d], VT[b,h,d,n] (bf16).
__global__ void __launch_bounds__(256) gemm_qkv_rope(
    const bf16* __restrict__ X, const bf16* __restrict__ WT,
    const float* __restrict__ freqs,
    bf16* __restrict__ Qb, bf16* __restrict__ Kb, bf16* __restrict__ VTb) {
  constexpr int KD = 2048;
  __shared__ __align__(16) bf16 Ash[128 * 72];
  __shared__ __align__(16) bf16 Bsh[128 * 72];
  const int tid = threadIdx.x;
  const int lane = tid & 31, wid = tid >> 5;
  const int wm = wid & 3, wn = wid >> 2;
  const int m0 = blockIdx.y * 128, n0 = blockIdx.x * 128;

  v8f acc[2][4];
#pragma unroll
  for (int i = 0; i < 2; i++)
#pragma unroll
    for (int j = 0; j < 4; j++) acc[i][j] = vzero8();

  for (int k0 = 0; k0 < KD; k0 += 64) {
#pragma unroll
    for (int i = 0; i < 4; i++) {
      int c = tid + i * 256;
      int row = c >> 3, col = (c & 7) << 3;
      async_b128(lds_off(&Ash[row * 72 + col]),
                 &X[(size_t)(m0 + row) * KD + k0 + col]);
      async_b128(lds_off(&Bsh[row * 72 + col]),
                 &WT[(size_t)(n0 + row) * KD + k0 + col]);
    }
    wait_async0();
    __syncthreads();
#pragma unroll
    for (int kc = 0; kc < 64; kc += 32) {
      v16bf af[2], bfr[4];
#pragma unroll
      for (int tm = 0; tm < 2; tm++)
        af[tm] = load_a_frag(&Ash[(wm * 32 + tm * 16) * 72 + kc], 72);
#pragma unroll
      for (int tn = 0; tn < 4; tn++)
        bfr[tn] = load_b_frag(&Bsh[(wn * 64 + tn * 16) * 72 + kc], 72);
#pragma unroll
      for (int tm = 0; tm < 2; tm++)
#pragma unroll
        for (int tn = 0; tn < 4; tn++)
          acc[tm][tn] = wmma_bf(af[tm], bfr[tn], acc[tm][tn]);
    }
    __syncthreads();
  }

  // Epilogue: C element at (M = r + 8*(lane>=16), N = lane&15) of each tile.
  // RoPE pair (even d, odd d) sits on adjacent lanes -> one shfl_xor.
  const int hi = lane >> 4, nlo = lane & 15;
#pragma unroll
  for (int tm = 0; tm < 2; tm++)
#pragma unroll
    for (int tn = 0; tn < 4; tn++)
#pragma unroll
      for (int r = 0; r < 8; r++) {
        float val = acc[tm][tn][r];
        float pv = __shfl_xor(val, 1);
        int m = m0 + wm * 32 + tm * 16 + r + hi * 8;   // token index
        int c = n0 + wn * 64 + tn * 16 + nlo;          // qkv column
        int bi = m >> 11, npos = m & 2047;
        int sec = c >> 11, cr = c & 2047;
        int h = cr >> 7, d = cr & 127;
        if (sec == 2) {
          VTb[(((size_t)bi * 16 + h) * 128 + d) * 2048 + npos] = f2bf(val);
        } else {
          float cs = freqs[(npos * 64 + (d >> 1)) * 2 + 0];
          float sn = freqs[(npos * 64 + (d >> 1)) * 2 + 1];
          float o = (d & 1) ? (val * cs + pv * sn) : (val * cs - pv * sn);
          bf16 ob = f2bf(o);
          size_t idx = (((size_t)bi * 16 + h) * 2048 + npos) * 128 + d;
          if (sec == 0) Qb[idx] = ob; else Kb[idx] = ob;
        }
      }
}

// ------------------------------------------------- flash attention --------
// One block per (b,h, 128-query tile); 8 waves, 16 queries each.
__global__ void __launch_bounds__(256) flash_attn(
    const bf16* __restrict__ Qb, const bf16* __restrict__ Kb,
    const bf16* __restrict__ VTb, bf16* __restrict__ AO) {
  __shared__ __align__(16) bf16 Ksh[64 * 136];    // [key][d], d contiguous
  __shared__ __align__(16) bf16 VTsh[128 * 72];   // [d][key], key contiguous
  __shared__ __align__(16) bf16 Psh[8 * 16 * 72]; // per-wave P repack scratch
  const int tid = threadIdx.x, lane = tid & 31, w = tid >> 5;
  const int bh = blockIdx.y;            // b*16 + h
  const int q0 = blockIdx.x * 128;
  const int qrow = q0 + w * 16;
  const float scale = 0.08838834764831845f;       // 1/sqrt(128)

  // Resident Q A-fragments: 16 queries x 128 dh as 4 K-chunks of 32.
  v16bf qf[4];
  {
    const bf16* qp = Qb + ((size_t)bh * 2048 + qrow) * 128;
#pragma unroll
    for (int kc = 0; kc < 4; kc++) qf[kc] = load_a_frag(qp + kc * 32, 128);
  }

  float mrow[8], lrow[8];
  v8f o[8];
#pragma unroll
  for (int r = 0; r < 8; r++) { mrow[r] = -1e30f; lrow[r] = 0.0f; }
#pragma unroll
  for (int t = 0; t < 8; t++) o[t] = vzero8();

  bf16* Pw = &Psh[w * 16 * 72];
  const int hi = lane >> 4, nlo = lane & 15;

  for (int kt = 0; kt < 2048; kt += 64) {
#pragma unroll
    for (int i = 0; i < 4; i++) {
      int c = tid + i * 256;
      { int r = c >> 4, col = (c & 15) << 3;      // 64 rows x 128 d
        async_b128(lds_off(&Ksh[r * 136 + col]),
                   &Kb[((size_t)bh * 2048 + kt + r) * 128 + col]); }
      { int r = c >> 3, col = (c & 7) << 3;       // 128 rows x 64 keys
        async_b128(lds_off(&VTsh[r * 72 + col]),
                   &VTb[((size_t)bh * 128 + r) * 2048 + kt + col]); }
    }
    wait_async0();
    __syncthreads();

    // S = scale * q . k  (16 queries x 64 keys)
    float s[4][8];
#pragma unroll
    for (int j = 0; j < 4; j++) {
      v8f sv = vzero8();
#pragma unroll
      for (int kc = 0; kc < 4; kc++) {
        v16bf bb = load_b_frag(&Ksh[(j * 16) * 136 + kc * 32], 136);
        sv = wmma_bf(qf[kc], bb, sv);
      }
#pragma unroll
      for (int r = 0; r < 8; r++) s[j][r] = sv[r] * scale;
    }

    // Online softmax: each half-wave (16 lanes) holds one row per VGPR r;
    // xor-1/2/4/8 reduces within the half (rows are disjoint across halves).
#pragma unroll
    for (int r = 0; r < 8; r++) {
      float mx = fmaxf(fmaxf(s[0][r], s[1][r]), fmaxf(s[2][r], s[3][r]));
      mx = fmaxf(mx, __shfl_xor(mx, 1));
      mx = fmaxf(mx, __shfl_xor(mx, 2));
      mx = fmaxf(mx, __shfl_xor(mx, 4));
      mx = fmaxf(mx, __shfl_xor(mx, 8));
      float mnew = fmaxf(mrow[r], mx);
      float alpha = __expf(mrow[r] - mnew);
      mrow[r] = mnew;
      float ps = 0.0f;
#pragma unroll
      for (int j = 0; j < 4; j++) { s[j][r] = __expf(s[j][r] - mnew); ps += s[j][r]; }
      ps += __shfl_xor(ps, 1);
      ps += __shfl_xor(ps, 2);
      ps += __shfl_xor(ps, 4);
      ps += __shfl_xor(ps, 8);
      lrow[r] = lrow[r] * alpha + ps;
#pragma unroll
      for (int t = 0; t < 8; t++) o[t][r] *= alpha;
    }

    // Repack P (C-layout) -> A-fragment layout via per-wave LDS scratch.
#pragma unroll
    for (int j = 0; j < 4; j++)
#pragma unroll
      for (int r = 0; r < 8; r++)
        Pw[(r + hi * 8) * 72 + j * 16 + nlo] = f2bf(s[j][r]);

    v16bf pf[2];
#pragma unroll
    for (int kc = 0; kc < 2; kc++) pf[kc] = load_a_frag(Pw + kc * 32, 72);

    // O += P . V  (B-fragments from VT: contiguous keys per dh column)
#pragma unroll
    for (int t = 0; t < 8; t++)
#pragma unroll
      for (int kc = 0; kc < 2; kc++) {
        v16bf bb = load_b_frag(&VTsh[(t * 16) * 72 + kc * 32], 72);
        o[t] = wmma_bf(pf[kc], bb, o[t]);
      }
    __syncthreads();
  }

  // Normalize and store to AO[b,n, h*128 + d] (bf16, GEMM2 input layout).
  const int b = bh >> 4, h = bh & 15;
#pragma unroll
  for (int r = 0; r < 8; r++) {
    float inv = 1.0f / lrow[r];
    int tok = b * 2048 + qrow + r + hi * 8;
#pragma unroll
    for (int t = 0; t < 8; t++)
      AO[(size_t)tok * 2048 + h * 128 + t * 16 + nlo] = f2bf(o[t][r] * inv);
  }
}

// ------------------------------------------------- GEMM2: out proj -------
__global__ void __launch_bounds__(256) gemm_out(
    const bf16* __restrict__ A, const bf16* __restrict__ WT,
    float* __restrict__ Out) {
  constexpr int KD = 2048;
  __shared__ __align__(16) bf16 Ash[128 * 72];
  __shared__ __align__(16) bf16 Bsh[128 * 72];
  const int tid = threadIdx.x;
  const int lane = tid & 31, wid = tid >> 5;
  const int wm = wid & 3, wn = wid >> 2;
  const int m0 = blockIdx.y * 128, n0 = blockIdx.x * 128;

  v8f acc[2][4];
#pragma unroll
  for (int i = 0; i < 2; i++)
#pragma unroll
    for (int j = 0; j < 4; j++) acc[i][j] = vzero8();

  for (int k0 = 0; k0 < KD; k0 += 64) {
#pragma unroll
    for (int i = 0; i < 4; i++) {
      int c = tid + i * 256;
      int row = c >> 3, col = (c & 7) << 3;
      async_b128(lds_off(&Ash[row * 72 + col]),
                 &A[(size_t)(m0 + row) * KD + k0 + col]);
      async_b128(lds_off(&Bsh[row * 72 + col]),
                 &WT[(size_t)(n0 + row) * KD + k0 + col]);
    }
    wait_async0();
    __syncthreads();
#pragma unroll
    for (int kc = 0; kc < 64; kc += 32) {
      v16bf af[2], bfr[4];
#pragma unroll
      for (int tm = 0; tm < 2; tm++)
        af[tm] = load_a_frag(&Ash[(wm * 32 + tm * 16) * 72 + kc], 72);
#pragma unroll
      for (int tn = 0; tn < 4; tn++)
        bfr[tn] = load_b_frag(&Bsh[(wn * 64 + tn * 16) * 72 + kc], 72);
#pragma unroll
      for (int tm = 0; tm < 2; tm++)
#pragma unroll
        for (int tn = 0; tn < 4; tn++)
          acc[tm][tn] = wmma_bf(af[tm], bfr[tn], acc[tm][tn]);
    }
    __syncthreads();
  }

  const int hi = lane >> 4, nlo = lane & 15;
#pragma unroll
  for (int tm = 0; tm < 2; tm++)
#pragma unroll
    for (int tn = 0; tn < 4; tn++)
#pragma unroll
      for (int r = 0; r < 8; r++) {
        int m = m0 + wm * 32 + tm * 16 + r + hi * 8;
        int c = n0 + wn * 64 + tn * 16 + nlo;
        Out[(size_t)m * 2048 + c] = acc[tm][tn][r];
      }
}

// ---------------------------------------------------------------- host ---
extern "C" void kernel_launch(void* const* d_in, const int* in_sizes, int n_in,
                              void* d_out, int out_size, void* d_ws, size_t ws_size,
                              hipStream_t stream) {
  (void)in_sizes; (void)n_in; (void)out_size; (void)ws_size;
  const float* x     = (const float*)d_in[0];  // [2,2048,2048]
  const float* freqs = (const float*)d_in[1];  // [2048,64,2]
  const float* wqkv  = (const float*)d_in[2];  // [2048,6144]
  const float* wo    = (const float*)d_in[3];  // [2048,2048]
  float* out = (float*)d_out;                  // [2,2048,2048]

  char* p = (char*)d_ws;
  bf16* xbf   = (bf16*)p; p += (size_t)4096 * 2048 * 2;
  bf16* wqkvT = (bf16*)p; p += (size_t)6144 * 2048 * 2;
  bf16* woT   = (bf16*)p; p += (size_t)2048 * 2048 * 2;
  bf16* Qb    = (bf16*)p; p += (size_t)32 * 2048 * 128 * 2;
  bf16* Kb    = (bf16*)p; p += (size_t)32 * 2048 * 128 * 2;
  bf16* VTb   = (bf16*)p; p += (size_t)32 * 128 * 2048 * 2;
  bf16* AO    = (bf16*)p; p += (size_t)4096 * 2048 * 2;   // ~117 MB total

  cast_f32_to_bf16<<<(4096 * 2048 / 4 + 255) / 256, 256, 0, stream>>>(
      x, xbf, 4096 * 2048);
  transpose_cast<<<dim3(6144 / 32, 2048 / 32), 256, 0, stream>>>(
      wqkv, wqkvT, 2048, 6144);
  transpose_cast<<<dim3(2048 / 32, 2048 / 32), 256, 0, stream>>>(
      wo, woT, 2048, 2048);
  gemm_qkv_rope<<<dim3(48, 32), 256, 0, stream>>>(
      xbf, wqkvT, freqs, Qb, Kb, VTb);
  flash_attn<<<dim3(16, 32), 256, 0, stream>>>(Qb, Kb, VTb, AO);
  gemm_out<<<dim3(16, 32), 256, 0, stream>>>(AO, woT, out);
}
